// SPN_18906446037565
// MI455X (gfx1250) — compile-verified, matrix operand order checked by reading, not run on hardware
//
#include <hip/hip_runtime.h>

// ---------------------------------------------------------------------------
// SPN forward on gfx1250: two bf16 WMMA GEMMs with fused bias/ReLU.
//   B=16384, X=256, N=2048, OS=256, HID=N-OS=1792
//   kernel1: z = x @ w0^T + b ; hidden -> relu -> bf16 ws ; out slice -> d_out
//   kernel2: d_out += h @ w1^T   (A staged via global_load_async_to_lds_b128)
// WG tile 128x256, 8 waves (2x4), wave tile 64x64 -> 16 WMMA per 16 ds_load_b128.
// ---------------------------------------------------------------------------

typedef __attribute__((ext_vector_type(16))) __bf16 v16bf;
typedef __attribute__((ext_vector_type(8)))  __bf16 v8bf;
typedef __attribute__((ext_vector_type(4)))  __bf16 v4bf;
typedef __attribute__((ext_vector_type(8)))  float  v8f;
typedef __attribute__((ext_vector_type(4)))  float  v4f;

#define BDIM   16384
#define XDIM   256
#define NDIM   2048
#define OSDIM  256
#define HID    1792            // NDIM - OSDIM (multiple of 256)
#define KC     64              // K chunk staged in LDS
#define LDA    72              // padded LDS row stride (bf16): 144B/row -> conflict-free

__device__ __forceinline__ __bf16 f2bf(float f) {
    unsigned u = __builtin_bit_cast(unsigned, f);
    unsigned r = u + 0x7FFFu + ((u >> 16) & 1u);   // round-to-nearest-even
    unsigned short hsh = (unsigned short)(r >> 16);
    return __builtin_bit_cast(__bf16, hsh);
}

// A/B fragment load from LDS per CDNA5 16-bit 16x32 layout:
//   lanes 0-15 : row = base+lane,     K = {ks+0..7, ks+16..23}
//   lanes 16-31: row = base+lane-16,  K = {ks+8..15, ks+24..31}
__device__ __forceinline__ v16bf load_frag(const __bf16* lds, int rowbase, int ks, int lane) {
    int row = rowbase + (lane & 15);
    int k0  = ks + ((lane & 16) ? 8 : 0);
    const __bf16* p = lds + row * LDA + k0;
    v8bf lo = *(const v8bf*)(p);        // ds_load_b128
    v8bf hi = *(const v8bf*)(p + 16);   // ds_load_b128
    v16bf f;
#pragma unroll
    for (int i = 0; i < 8; ++i) { f[i] = lo[i]; f[i + 8] = hi[i]; }
    return f;
}

// ---------------------------------------------------------------------------
// Kernel 1: z = x @ w0^T + bias. grid = (128 Mtiles, 8 Ntiles), 256 threads.
// ---------------------------------------------------------------------------
__global__ __launch_bounds__(256)
void spn_gemm1(const float* __restrict__ x, const float* __restrict__ w0,
               const float* __restrict__ bias, float* __restrict__ zout,
               __bf16* __restrict__ hout) {
    __shared__ __bf16 lA[128 * LDA];   // x tile:  128 rows x 64 k
    __shared__ __bf16 lB[256 * LDA];   // w0 tile: 256 rows x 64 k

    const int t    = threadIdx.x;
    const int lane = t & 31;
    const int wave = t >> 5;
    const int wm   = wave >> 2;           // 0..1  (64 rows each)
    const int wn   = wave & 3;            // 0..3  (64 cols each)
    const int rbase = blockIdx.x * 128;   // batch rows
    const int cbase = blockIdx.y * 256;   // node columns

    v8f acc[4][4];
#pragma unroll
    for (int a = 0; a < 4; ++a)
#pragma unroll
        for (int b = 0; b < 4; ++b) acc[a][b] = (v8f)0.f;

    for (int kc = 0; kc < XDIM; kc += KC) {
        // stage A: x[rbase..+128, kc..+64] f32 -> bf16   (2048 float4 / 256 thr)
#pragma unroll
        for (int i = 0; i < 8; ++i) {
            int elem = (t + i * 256) * 4;
            int row = elem >> 6;
            int col = elem & 63;
            v4f v = *(const v4f*)(x + (size_t)(rbase + row) * XDIM + kc + col);
            v4bf o; o[0] = f2bf(v[0]); o[1] = f2bf(v[1]); o[2] = f2bf(v[2]); o[3] = f2bf(v[3]);
            *(v4bf*)(lA + row * LDA + col) = o;
        }
        // stage B: w0[cbase..+256, kc..+64] f32 -> bf16  (4096 float4 / 256 thr)
#pragma unroll
        for (int i = 0; i < 16; ++i) {
            int elem = (t + i * 256) * 4;
            int row = elem >> 6;
            int col = elem & 63;
            v4f v = *(const v4f*)(w0 + (size_t)(cbase + row) * XDIM + kc + col);
            v4bf o; o[0] = f2bf(v[0]); o[1] = f2bf(v[1]); o[2] = f2bf(v[2]); o[3] = f2bf(v[3]);
            *(v4bf*)(lB + row * LDA + col) = o;
        }
        __syncthreads();
#pragma unroll
        for (int ks = 0; ks < KC; ks += 32) {
            v16bf af[4], bf[4];
#pragma unroll
            for (int tm = 0; tm < 4; ++tm) af[tm] = load_frag(lA, wm * 64 + tm * 16, ks, lane);
#pragma unroll
            for (int tn = 0; tn < 4; ++tn) bf[tn] = load_frag(lB, wn * 64 + tn * 16, ks, lane);
#pragma unroll
            for (int tm = 0; tm < 4; ++tm)
#pragma unroll
                for (int tn = 0; tn < 4; ++tn)
                    acc[tm][tn] = __builtin_amdgcn_wmma_f32_16x16x32_bf16(
                        false, af[tm], false, bf[tn], (short)0, acc[tm][tn], false, false);
        }
        __syncthreads();
    }

    // epilogue: C layout -> N = lane&15 ; M = vgpr + 8*(lane>=16)
    const int nloc  = lane & 15;
    const int mhalf = (lane >> 4) << 3;
#pragma unroll
    for (int tn = 0; tn < 4; ++tn) {
        int col = cbase + wn * 64 + tn * 16 + nloc;
        float bv = bias[col];
        if (col < HID) {                               // hidden: bias + relu -> bf16 ws
#pragma unroll
            for (int tm = 0; tm < 4; ++tm)
#pragma unroll
                for (int v = 0; v < 8; ++v) {
                    int m = rbase + wm * 64 + tm * 16 + mhalf + v;
                    float z = acc[tm][tn][v] + bv;
                    z = z > 0.f ? z : 0.f;
                    hout[(size_t)m * HID + col] = f2bf(z);
                }
        } else {                                       // output slice: bias -> f32 d_out
            int co = col - HID;
#pragma unroll
            for (int tm = 0; tm < 4; ++tm)
#pragma unroll
                for (int v = 0; v < 8; ++v) {
                    int m = rbase + wm * 64 + tm * 16 + mhalf + v;
                    zout[(size_t)m * OSDIM + co] = acc[tm][tn][v] + bv;
                }
        }
    }
}

// ---------------------------------------------------------------------------
// Kernel 2: out += h @ w1^T. grid = (128 Mtiles, 1), 256 threads.
// A staged with GLOBAL_LOAD_ASYNC_TO_LDS_B128 (ASYNCcnt path).
// ---------------------------------------------------------------------------
__global__ __launch_bounds__(256)
void spn_gemm2(const __bf16* __restrict__ h, const float* __restrict__ w1,
               float* __restrict__ out) {
    __shared__ __bf16 lA[128 * LDA];   // h tile:  128 rows x 64 k (bf16 copy)
    __shared__ __bf16 lB[256 * LDA];   // w1 tile: 256 rows x 64 k (f32 -> bf16)

    const int t    = threadIdx.x;
    const int lane = t & 31;
    const int wave = t >> 5;
    const int wm   = wave >> 2;
    const int wn   = wave & 3;
    const int rbase = blockIdx.x * 128;

    v8f acc[4][4];
#pragma unroll
    for (int a = 0; a < 4; ++a)
#pragma unroll
        for (int b = 0; b < 4; ++b) acc[a][b] = (v8f)0.f;

    for (int kc = 0; kc < HID; kc += KC) {
        // stage A: async DMA h[rbase..+128, kc..+64] -> LDS (1024 x 16B / 256 thr)
#pragma unroll
        for (int i = 0; i < 4; ++i) {
            int elem = (t + i * 256) * 8;
            int row = elem >> 6;
            int col = elem & 63;
            // low 32 bits of a generic pointer into LDS == wave-relative LDS byte addr
            unsigned ldsoff = (unsigned)(size_t)(lA + row * LDA + col);
            const __bf16* gp = h + (size_t)(rbase + row) * HID + kc + col;
            asm volatile("global_load_async_to_lds_b128 %0, %1, off"
                         :: "v"(ldsoff), "v"(gp) : "memory");
        }
        // stage B: w1[0..256, kc..+64] f32 -> bf16
#pragma unroll
        for (int i = 0; i < 16; ++i) {
            int elem = (t + i * 256) * 4;
            int row = elem >> 6;
            int col = elem & 63;
            v4f v = *(const v4f*)(w1 + (size_t)row * HID + kc + col);
            v4bf o; o[0] = f2bf(v[0]); o[1] = f2bf(v[1]); o[2] = f2bf(v[2]); o[3] = f2bf(v[3]);
            *(v4bf*)(lB + row * LDA + col) = o;
        }
        asm volatile("s_wait_asynccnt 0" ::: "memory");
        __syncthreads();
#pragma unroll
        for (int ks = 0; ks < KC; ks += 32) {
            v16bf af[4], bf[4];
#pragma unroll
            for (int tm = 0; tm < 4; ++tm) af[tm] = load_frag(lA, wm * 64 + tm * 16, ks, lane);
#pragma unroll
            for (int tn = 0; tn < 4; ++tn) bf[tn] = load_frag(lB, wn * 64 + tn * 16, ks, lane);
#pragma unroll
            for (int tm = 0; tm < 4; ++tm)
#pragma unroll
                for (int tn = 0; tn < 4; ++tn)
                    acc[tm][tn] = __builtin_amdgcn_wmma_f32_16x16x32_bf16(
                        false, af[tm], false, bf[tn], (short)0, acc[tm][tn], false, false);
        }
        __syncthreads();
    }

    const int nloc  = lane & 15;
    const int mhalf = (lane >> 4) << 3;
#pragma unroll
    for (int tn = 0; tn < 4; ++tn) {
        int col = wn * 64 + tn * 16 + nloc;           // 0..255
#pragma unroll
        for (int tm = 0; tm < 4; ++tm)
#pragma unroll
            for (int v = 0; v < 8; ++v) {
                int m = rbase + wm * 64 + tm * 16 + mhalf + v;
                size_t idx = (size_t)m * OSDIM + col;
                out[idx] += acc[tm][tn][v];   // z_out written by kernel1
            }
    }
}

// ---------------------------------------------------------------------------
extern "C" void kernel_launch(void* const* d_in, const int* in_sizes, int n_in,
                              void* d_out, int out_size, void* d_ws, size_t ws_size,
                              hipStream_t stream) {
    (void)in_sizes; (void)n_in; (void)out_size; (void)ws_size;
    const float* x    = (const float*)d_in[0];   // [B, X]
    const float* w0   = (const float*)d_in[1];   // [N, X]
    const float* w1   = (const float*)d_in[2];   // [OS, HID]
    const float* bias = (const float*)d_in[3];   // [N]
    float*  out = (float*)d_out;                 // [B, OS]
    __bf16* hws = (__bf16*)d_ws;                 // [B, HID] bf16 intermediate (58.7 MB)

    dim3 blk(256);
    dim3 g1(BDIM / 128, NDIM / 256);   // 128 x 8
    dim3 g2(BDIM / 128, 1);            // 128 x 1
    spn_gemm1<<<g1, blk, 0, stream>>>(x, w0, bias, out, hws);
    spn_gemm2<<<g2, blk, 0, stream>>>(hws, w1, out);
}
